// LinearMoELayer_18176301597482
// MI455X (gfx1250) — matrix operand order repaired and checked by reading, not compiled, not run on hardware
//
#include <hip/hip_runtime.h>
#include <hip/hip_bf16.h>
#include <math.h>

// MoE: B=2, S=2048 -> T=4096 tokens, D_IN=D_OUT=4096, E=8 experts, top-2.
#define T_TOK 4096
#define D     4096
#define NE    8
#define KT    128    // K elements (bf16) staged in LDS per iteration
#define MT    32     // tokens per block tile
#define NTILE 128    // output columns per block (8 waves x 16)

typedef __attribute__((ext_vector_type(16))) __bf16 v16bf;
typedef __attribute__((ext_vector_type(4)))  __bf16 v4bf;
typedef __attribute__((ext_vector_type(8)))  float  v8f;
typedef __attribute__((ext_vector_type(4)))  int    v4i;

#if defined(__HIP_DEVICE_COMPILE__) && __has_builtin(__builtin_amdgcn_global_load_async_to_lds_b128)
#define HAVE_ASYNC_LDS 1
#else
#define HAVE_ASYNC_LDS 0
#endif

__device__ __forceinline__ __bf16 f2bf(float f) {
    unsigned u = __float_as_uint(f);
    u += 0x7FFFu + ((u >> 16) & 1u);      // round-to-nearest-even
    unsigned short h = (unsigned short)(u >> 16);
    return __builtin_bit_cast(__bf16, h);
}

// 16-byte global->LDS copy: async DMA on gfx1250 when available.
__device__ __forceinline__ void cp16_g2l(const __bf16* gsrc, __bf16* ldst) {
#if HAVE_ASYNC_LDS
    __builtin_amdgcn_global_load_async_to_lds_b128(
        (__attribute__((address_space(1))) v4i*)gsrc,
        (__attribute__((address_space(3))) v4i*)ldst, 0, 0);
#else
    *(uint4*)ldst = *(const uint4*)gsrc;
#endif
}

__device__ __forceinline__ void wait_async_lds() {
#if HAVE_ASYNC_LDS
#if __has_builtin(__builtin_amdgcn_s_wait_asynccnt)
    __builtin_amdgcn_s_wait_asynccnt(0);
#else
    asm volatile("s_wait_asynccnt 0x0" ::: "memory");
#endif
#endif
}

// ---------------- init: zero routing state ----------------
__global__ void moe_init_kernel(int* counts, float* importance) {
    int i = threadIdx.x;
    if (i < NE) { counts[i] = 0; importance[i] = 0.0f; }
}

// ---------------- one-shot f32 -> bf16 conversion into workspace ----------------
__global__ void cvt_bf16_kernel(const float* __restrict__ src, __bf16* __restrict__ dst, long n4) {
    long i = (long)blockIdx.x * blockDim.x + threadIdx.x;
    long stride = (long)gridDim.x * blockDim.x;
    for (; i < n4; i += stride) {
        float4 v = ((const float4*)src)[i];
        v4bf o;
        o[0] = f2bf(v.x); o[1] = f2bf(v.y); o[2] = f2bf(v.z); o[3] = f2bf(v.w);
        ((v4bf*)dst)[i] = o;
    }
}

// ---------------- gate: logits, top-2, softmax, routing lists, bias seed ----------------
__global__ void moe_gate_kernel(const float* __restrict__ x,
                                const float* __restrict__ gw1,   // [E, D]
                                const float* __restrict__ gw2,   // [E, E]
                                const float* __restrict__ eb,    // [E, D]
                                float* __restrict__ y,           // [T, D]
                                int* __restrict__ counts,
                                float* __restrict__ importance,
                                int* __restrict__ tok_idx,       // [E, T]
                                float* __restrict__ tok_score) { // [E, T]
    const int t = blockIdx.x;
    const int tid = threadIdx.x;          // 128 threads
    __shared__ float s_red[NE][128];
    __shared__ float s_h[NE];
    __shared__ float s_log[NE];
    __shared__ int   s_e[2];
    __shared__ float s_s[2];

    float p[NE];
#pragma unroll
    for (int e = 0; e < NE; ++e) p[e] = 0.0f;
    const float* xr = x + (size_t)t * D;
    for (int i = tid; i < D; i += 128) {
        float xv = xr[i];
#pragma unroll
        for (int e = 0; e < NE; ++e) p[e] += xv * gw1[e * D + i];
    }
#pragma unroll
    for (int e = 0; e < NE; ++e) s_red[e][tid] = p[e];
    __syncthreads();
    if (tid < NE) {
        float s = 0.0f;
        for (int j = 0; j < 128; ++j) s += s_red[tid][j];
        s_h[tid] = tanhf(s);
    }
    __syncthreads();
    if (tid < NE) {
        float l = 0.0f;
#pragma unroll
        for (int e = 0; e < NE; ++e) l += s_h[e] * gw2[tid * NE + e];
        s_log[tid] = l;
    }
    __syncthreads();
    if (tid == 0) {
        // top-2 with lower-index tie-break (matches jax.lax.top_k)
        int i0 = 0; float l0 = s_log[0];
        for (int j = 1; j < NE; ++j) if (s_log[j] > l0) { l0 = s_log[j]; i0 = j; }
        int i1 = (i0 == 0) ? 1 : 0; float l1 = s_log[i1];
        for (int j = 0; j < NE; ++j) { if (j == i0 || j == i1) continue; if (s_log[j] > l1) { l1 = s_log[j]; i1 = j; } }
        float e1 = __expf(l1 - l0);
        float s0 = 1.0f / (1.0f + e1);
        float s1 = e1 * s0;
        int p0 = atomicAdd(&counts[i0], 1);
        tok_idx[i0 * T_TOK + p0] = t; tok_score[i0 * T_TOK + p0] = s0;
        int p1 = atomicAdd(&counts[i1], 1);
        tok_idx[i1 * T_TOK + p1] = t; tok_score[i1 * T_TOK + p1] = s1;
        atomicAdd(&importance[i0], s0);
        atomicAdd(&importance[i1], s1);
        s_e[0] = i0; s_e[1] = i1; s_s[0] = s0; s_s[1] = s1;
    }
    __syncthreads();
    const int   e0 = s_e[0], e1x = s_e[1];
    const float s0 = s_s[0], s1 = s_s[1];
    float* yr = y + (size_t)t * D;
    const float* b0 = eb + (size_t)e0  * D;
    const float* b1 = eb + (size_t)e1x * D;
    for (int o = tid; o < D; o += 128) yr[o] = s0 * b0[o] + s1 * b1[o];
}

// ---------------- grouped expert GEMM: bf16 WMMA, async LDS staging, scatter-add ----------------
__global__ void __launch_bounds__(256)
moe_expert_gemm(const __bf16* __restrict__ xb,    // [T, D] bf16
                const __bf16* __restrict__ wb,    // [E, D, D] bf16 row-major (out, in)
                float* __restrict__ y,            // [T, D]
                const int* __restrict__ counts,
                const int* __restrict__ tok_idx,
                const float* __restrict__ tok_score) {
    const int e     = blockIdx.z;
    const int mtile = blockIdx.y;
    const int nbase = blockIdx.x * NTILE;
    const int cnt   = counts[e];
    if (mtile * MT >= cnt) return;        // unused capacity tile for this expert

    const int tid  = threadIdx.x;         // 256 threads = 8 waves (wave32)
    const int lane = tid & 31;
    const int wave = tid >> 5;            // wave owns 16 output cols
    const int col  = lane & 15;
    const int half = lane >> 4;

    __shared__ __bf16 As[MT][KT];         // 8 KB  gathered token tile
    __shared__ __bf16 Bs[NTILE][KT];      // 32 KB weight tile
    __shared__ int    s_tok[MT];
    __shared__ float  s_sc[MT];

    if (tid < MT) {
        int idx = mtile * MT + tid;
        if (idx < cnt) { s_tok[tid] = tok_idx[e * T_TOK + idx]; s_sc[tid] = tok_score[e * T_TOK + idx]; }
        else           { s_tok[tid] = 0;                        s_sc[tid] = 0.0f; }
    }
    __syncthreads();

    const __bf16* W = wb + (size_t)e * D * D;
    v8f acc0 = {}, acc1 = {};
    const int SEG = KT / 8;               // 16-byte segments per row

    for (int k0 = 0; k0 < D; k0 += KT) {
        // stage A: MT x KT bf16 = 512 segs, 2 per thread (gathered rows)
        for (int s = tid; s < MT * SEG; s += 256) {
            int r = s / SEG, c = s % SEG;
            cp16_g2l(xb + (size_t)s_tok[r] * D + k0 + c * 8, &As[r][c * 8]);
        }
        // stage B: NTILE x KT bf16 = 2048 segs, 8 per thread
        for (int s = tid; s < NTILE * SEG; s += 256) {
            int r = s / SEG, c = s % SEG;
            cp16_g2l(W + (size_t)(nbase + r) * D + k0 + c * 8, &Bs[r][c * 8]);
        }
        if (k0 + KT < D) {  // prefetch next weight stage (global_prefetch_b8 path)
            __builtin_prefetch(W + (size_t)(nbase + (tid >> 1)) * D + k0 + KT, 0, 1);
        }
        wait_async_lds();
        __syncthreads();

#pragma unroll
        for (int ks = 0; ks < KT; ks += 32) {
            v16bf a0, a1, b;
#pragma unroll
            for (int v = 0; v < 8; ++v) {
                // A 16x32 bf16 layout: lanes 0-15 M=0..15; VGPR v holds a K pair
                int kb = ks + (v & 3) * 2 + half * 8 + (v >> 2) * 16;
                a0[2 * v]     = As[col][kb];
                a0[2 * v + 1] = As[col][kb + 1];
                a1[2 * v]     = As[16 + col][kb];
                a1[2 * v + 1] = As[16 + col][kb + 1];
                // B 32x16 bf16 layout: lane%16 = column, K order matches A
                int kc = ks + (v & 3) * 2 + half * 8 + (v >> 2) * 16;
                b[2 * v]     = Bs[wave * 16 + col][kc];
                b[2 * v + 1] = Bs[wave * 16 + col][kc + 1];
            }
            acc0 = __builtin_amdgcn_wmma_f32_16x16x32_bf16(false, a0, false, b,
                                                           (short)0, acc0, false, false);
            acc1 = __builtin_amdgcn_wmma_f32_16x16x32_bf16(false, a1, false, b,
                                                           (short)0, acc1, false, false);
        }
        __syncthreads();
    }

    // epilogue: C layout -> VGPR r, lanes 0-15: M=r; lanes 16-31: M=r+8; N=lane&15
    const int ncol = nbase + wave * 16 + col;
#pragma unroll
    for (int r = 0; r < 8; ++r) {
        int m = r + 8 * half;
        float s0 = s_sc[m];
        if (s0 != 0.0f) atomicAdd(&y[(size_t)s_tok[m] * D + ncol], acc0[r] * s0);
        float s1 = s_sc[m + 16];
        if (s1 != 0.0f) atomicAdd(&y[(size_t)s_tok[m + 16] * D + ncol], acc1[r] * s1);
    }
}

// ---------------- balance loss (scalar) ----------------
__global__ void moe_balance_kernel(const int* counts, const float* importance, float* loss_out) {
    if (threadIdx.x != 0 || blockIdx.x != 0) return;
    float mi = 0.0f, ml = 0.0f;
    for (int e = 0; e < NE; ++e) { mi += importance[e]; ml += (float)counts[e]; }
    mi *= (1.0f / NE); ml *= (1.0f / NE);
    float vi = 0.0f, vl = 0.0f;
    for (int e = 0; e < NE; ++e) {
        float di = importance[e] - mi; vi += di * di;
        float dl = (float)counts[e] - ml; vl += dl * dl;
    }
    vi *= (1.0f / (NE - 1)); vl *= (1.0f / (NE - 1));   // unbiased var (ddof=1)
    const float eps = 1e-10f;
    *loss_out = (vi / (mi * mi + eps) + vl / (ml * ml + eps)) * 0.01f;
}

extern "C" void kernel_launch(void* const* d_in, const int* in_sizes, int n_in,
                              void* d_out, int out_size, void* d_ws, size_t ws_size,
                              hipStream_t stream) {
    (void)in_sizes; (void)n_in; (void)out_size; (void)ws_size;
    const float* x   = (const float*)d_in[0];   // [2,2048,4096]
    const float* gw1 = (const float*)d_in[1];   // [8,4096]
    const float* gw2 = (const float*)d_in[2];   // [8,8]
    const float* ew  = (const float*)d_in[3];   // [8,4096,4096]
    const float* eb  = (const float*)d_in[4];   // [8,4096]
    float* y = (float*)d_out;                   // [T*D] + 1 loss scalar

    char* ws = (char*)d_ws;
    int*    counts     = (int*)ws;                                    // 8 ints
    float*  importance = (float*)(ws + 64);                           // 8 floats
    int*    tok_idx    = (int*)(ws + 128);                            // [E,T]
    float*  tok_score  = (float*)(ws + 128 + (size_t)NE * T_TOK * 4); // [E,T]
    __bf16* xb = (__bf16*)(ws + (1u << 20));                          // 32 MB  [T,D]
    __bf16* wbf = (__bf16*)(ws + (1u << 20) + (size_t)T_TOK * D * 2); // 256 MB [E,D,D]

    moe_init_kernel<<<1, 32, 0, stream>>>(counts, importance);
    cvt_bf16_kernel<<<2048, 256, 0, stream>>>(x,  xb,  (long)T_TOK * D / 4);
    cvt_bf16_kernel<<<8192, 256, 0, stream>>>(ew, wbf, (long)NE * D * D / 4);
    moe_gate_kernel<<<T_TOK, 128, 0, stream>>>(x, gw1, gw2, eb, y,
                                               counts, importance, tok_idx, tok_score);
    dim3 grid(D / NTILE, T_TOK / MT, NE);
    moe_expert_gemm<<<grid, 256, 0, stream>>>(xb, wbf, y, counts, tok_idx, tok_score);
    moe_balance_kernel<<<1, 32, 0, stream>>>(counts, importance, y + (size_t)T_TOK * D);
}
